// MultiLayerLSTMCells_23742579212831
// MI455X (gfx1250) — matrix operand order repaired and checked by reading, not compiled, not run on hardware
//
#include <hip/hip_runtime.h>
#include <math.h>

typedef __attribute__((ext_vector_type(16))) __bf16 v16bf;
typedef __attribute__((ext_vector_type(8)))  __bf16 v8bf;
typedef __attribute__((ext_vector_type(4)))  __bf16 v4bf;
typedef __attribute__((ext_vector_type(8)))  float  v8f;

#define NLAYERS 4
#define BATCH   256
#define HID     1024
#define KDIM    1024
#define BM      32     // batch tile per WG
#define BN      32     // hidden tile per WG (x4 gates)
#define BK      32     // K slab per stage
#define LDSW    40     // padded row stride (bf16 elems): conflict-free, 16B aligned

// One LSTM layer: gates = inp*w_ih^T + hprev*w_hh^T + b; then cell update.
// Grid: (BATCH/BM, HID/BN), 128 threads (4 wave32), each wave owns a 16x16
// tile for each of the 4 gates (i,f,g,o). Software-pipelined K loop with
// double-buffered LDS: global loads for slab k+1 issue before the WMMAs of
// slab k, conversion/stores land after, one barrier per iteration.
__global__ __launch_bounds__(128) void lstm_layer_kernel(
    const float* __restrict__ inp,    // (BATCH, KDIM)
    const float* __restrict__ hprev,  // (BATCH, HID)   = h0[layer]
    const float* __restrict__ cprev,  // (BATCH, HID)   = c0[layer]
    const float* __restrict__ w_ih,   // (4*HID, KDIM)  row-major
    const float* __restrict__ w_hh,   // (4*HID, HID)
    const float* __restrict__ b_ih,   // (4*HID)
    const float* __restrict__ b_hh,   // (4*HID)
    float* __restrict__ h_out,        // (BATCH, HID)
    float* __restrict__ c_out)        // (BATCH, HID)
{
    // bf16 hi/lo split staging, double buffered
    __shared__ __bf16 AsHi[2][BM * LDSW];
    __shared__ __bf16 AsLo[2][BM * LDSW];
    __shared__ __bf16 BsHi[2][4 * BN * LDSW];
    __shared__ __bf16 BsLo[2][4 * BN * LDSW];

    const int tid  = threadIdx.x;
    const int wave = tid >> 5;
    const int lane = tid & 31;
    const int la   = lane & 15;
    const int hiK  = (lane >> 4) & 1;   // lane group (K-half selector)
    const int wm   = wave & 1;          // M sub-tile (0..1)
    const int wn   = (wave >> 1) & 1;   // N sub-tile (0..1)
    const int gm0  = blockIdx.x * BM;
    const int gn0  = blockIdx.y * BN;

    v8f acc[4];
    {
        v8f z = {0.f, 0.f, 0.f, 0.f, 0.f, 0.f, 0.f, 0.f};
        #pragma unroll
        for (int g = 0; g < 4; ++g) acc[g] = z;
    }

    // register staging for the in-flight global slab
    float4 aReg[2];
    float4 bReg[8];

    auto global_fetch = [&](const float* Ag, const float* Wg, int k0) {
        #pragma unroll
        for (int i = 0; i < 2; ++i) {
            int idx = i * 128 + tid;            // 0..255 float4s
            int r   = idx >> 3;                 // 0..31
            int c4  = (idx & 7) << 2;
            aReg[i] = *(const float4*)&Ag[(size_t)(gm0 + r) * KDIM + k0 + c4];
        }
        #pragma unroll
        for (int i = 0; i < 8; ++i) {
            int idx = i * 128 + tid;            // 0..1023 float4s
            int rc  = idx >> 3;                 // 0..127 = g*32 + n
            int c4  = (idx & 7) << 2;
            int g   = rc >> 5;
            int n   = rc & 31;
            const float* src = &Wg[(size_t)(g * HID + gn0 + n) * KDIM + k0 + c4];
            bReg[i] = *(const float4*)src;
            if (k0 + 2 * BK < KDIM) __builtin_prefetch(src + 2 * BK, 0, 1);
        }
    };

    auto lds_store = [&](int buf) {
        #pragma unroll
        for (int i = 0; i < 2; ++i) {
            int idx = i * 128 + tid;
            int r   = idx >> 3;
            int c4  = (idx & 7) << 2;
            float f[4] = {aReg[i].x, aReg[i].y, aReg[i].z, aReg[i].w};
            v4bf hi, lo;
            #pragma unroll
            for (int e = 0; e < 4; ++e) {
                __bf16 h = (__bf16)f[e];
                hi[e] = h;
                lo[e] = (__bf16)(f[e] - (float)h);
            }
            *(v4bf*)&AsHi[buf][r * LDSW + c4] = hi;
            *(v4bf*)&AsLo[buf][r * LDSW + c4] = lo;
        }
        #pragma unroll
        for (int i = 0; i < 8; ++i) {
            int idx = i * 128 + tid;
            int rc  = idx >> 3;
            int c4  = (idx & 7) << 2;
            float f[4] = {bReg[i].x, bReg[i].y, bReg[i].z, bReg[i].w};
            v4bf hi, lo;
            #pragma unroll
            for (int e = 0; e < 4; ++e) {
                __bf16 h = (__bf16)f[e];
                hi[e] = h;
                lo[e] = (__bf16)(f[e] - (float)h);
            }
            *(v4bf*)&BsHi[buf][rc * LDSW + c4] = hi;
            *(v4bf*)&BsLo[buf][rc * LDSW + c4] = lo;
        }
    };

    auto compute = [&](int buf) {
        // A operand per ISA 16-bit A layout:
        // lanes 0-15:  e0..7 -> K0..7,   e8..15 -> K16..23
        // lanes 16-31: e0..7 -> K8..15,  e8..15 -> K24..31
        const __bf16* arH = &AsHi[buf][(wm * 16 + la) * LDSW];
        const __bf16* arL = &AsLo[buf][(wm * 16 + la) * LDSW];
        v8bf a0h = *(const v8bf*)&arH[hiK * 8];
        v8bf a1h = *(const v8bf*)&arH[16 + hiK * 8];
        v8bf a0l = *(const v8bf*)&arL[hiK * 8];
        v8bf a1l = *(const v8bf*)&arL[16 + hiK * 8];
        v16bf a_hi = __builtin_shufflevector(a0h, a1h,
            0,1,2,3,4,5,6,7,8,9,10,11,12,13,14,15);
        v16bf a_lo = __builtin_shufflevector(a0l, a1l,
            0,1,2,3,4,5,6,7,8,9,10,11,12,13,14,15);

        #pragma unroll
        for (int g = 0; g < 4; ++g) {
            // B layout: lane la = column N; same K order as A (e0..7 -> K 8*hiK+e, e8..15 -> K 16+8*hiK+e-8)
            const __bf16* brH = &BsHi[buf][(g * BN + wn * 16 + la) * LDSW];
            const __bf16* brL = &BsLo[buf][(g * BN + wn * 16 + la) * LDSW];
            v8bf b0h = *(const v8bf*)&brH[hiK * 8];
            v8bf b1h = *(const v8bf*)&brH[16 + hiK * 8];
            v8bf b0l = *(const v8bf*)&brL[hiK * 8];
            v8bf b1l = *(const v8bf*)&brL[16 + hiK * 8];
            v16bf b_hi = __builtin_shufflevector(b0h, b1h,
                0,1,2,3,4,5,6,7,8,9,10,11,12,13,14,15);
            v16bf b_lo = __builtin_shufflevector(b0l, b1l,
                0,1,2,3,4,5,6,7,8,9,10,11,12,13,14,15);

            // fp32-accurate split product: Ah*Bh + Al*Bh + Ah*Bl
            acc[g] = __builtin_amdgcn_wmma_f32_16x16x32_bf16(
                false, a_hi, false, b_hi, (short)0, acc[g], false, false);
            acc[g] = __builtin_amdgcn_wmma_f32_16x16x32_bf16(
                false, a_lo, false, b_hi, (short)0, acc[g], false, false);
            acc[g] = __builtin_amdgcn_wmma_f32_16x16x32_bf16(
                false, a_hi, false, b_lo, (short)0, acc[g], false, false);
        }
    };

    #pragma unroll
    for (int ph = 0; ph < 2; ++ph) {
        const float* Ag = ph ? hprev : inp;   // activation operand
        const float* Wg = ph ? w_hh  : w_ih;  // weight operand

        // pipeline prologue: slab 0 into buffer 0
        global_fetch(Ag, Wg, 0);
        __syncthreads();   // prior phase's tail reads complete before overwrite
        lds_store(0);
        __syncthreads();

        #pragma unroll 2
        for (int kc = 0; kc < KDIM / BK; ++kc) {
            if (kc < KDIM / BK - 1)
                global_fetch(Ag, Wg, (kc + 1) * BK);   // hide global latency
            compute(kc & 1);                            // WMMAs on current buf
            if (kc < KDIM / BK - 1)
                lds_store((kc + 1) & 1);                // fill other buf
            __syncthreads();
        }
    }

    // ---- fused LSTM cell epilogue ----
    // C/D layout: VGPR p -> M = p (lanes 0-15) or 8+p (lanes 16-31); N = lane&15
    const int j = gn0 + wn * 16 + la;       // hidden index
    float bias[4];
    #pragma unroll
    for (int g = 0; g < 4; ++g)
        bias[g] = b_ih[g * HID + j] + b_hh[g * HID + j];

    #pragma unroll
    for (int p = 0; p < 8; ++p) {
        int m = gm0 + wm * 16 + hiK * 8 + p;    // batch index
        float gi = acc[0][p] + bias[0];
        float gf = acc[1][p] + bias[1];
        float gg = acc[2][p] + bias[2];
        float go = acc[3][p] + bias[3];
        float is = 1.f / (1.f + __expf(-gi));
        float fs = 1.f / (1.f + __expf(-gf));
        float gt = tanhf(gg);
        float os = 1.f / (1.f + __expf(-go));
        float cold = cprev[(size_t)m * HID + j];
        float cn = fs * cold + is * gt;
        float hn = os * tanhf(cn);
        c_out[(size_t)m * HID + j] = cn;
        h_out[(size_t)m * HID + j] = hn;
    }
}

extern "C" void kernel_launch(void* const* d_in, const int* in_sizes, int n_in,
                              void* d_out, int out_size, void* d_ws, size_t ws_size,
                              hipStream_t stream) {
    (void)in_sizes; (void)n_in; (void)out_size; (void)d_ws; (void)ws_size;
    const float* x    = (const float*)d_in[0];
    const float* h0   = (const float*)d_in[1];
    const float* c0   = (const float*)d_in[2];
    const float* w_ih = (const float*)d_in[3];
    const float* w_hh = (const float*)d_in[4];
    const float* b_ih = (const float*)d_in[5];
    const float* b_hh = (const float*)d_in[6];

    float* hs = (float*)d_out;                       // (L,B,H)
    float* cs = hs + (size_t)NLAYERS * BATCH * HID;  // (L,B,H)

    dim3 grid(BATCH / BM, HID / BN);   // 8 x 32 = 256 workgroups
    const float* inp = x;
    for (int l = 0; l < NLAYERS; ++l) {
        lstm_layer_kernel<<<grid, 128, 0, stream>>>(
            inp,
            h0 + (size_t)l * BATCH * HID,
            c0 + (size_t)l * BATCH * HID,
            w_ih + (size_t)l * 4 * HID * KDIM,
            w_hh + (size_t)l * 4 * HID * HID,
            b_ih + (size_t)l * 4 * HID,
            b_hh + (size_t)l * 4 * HID,
            hs + (size_t)l * BATCH * HID,
            cs + (size_t)l * BATCH * HID);
        inp = hs + (size_t)l * BATCH * HID;   // chain layers through d_out
    }
}
